// ParallelAttention_49203145343016
// MI455X (gfx1250) — matrix operand order repaired and checked by reading, not compiled, not run on hardware
//
#include <hip/hip_runtime.h>
#include <hip/hip_bf16.h>

typedef __attribute__((ext_vector_type(16))) __bf16 v16bf;
typedef __attribute__((ext_vector_type(8)))  float  v8f;

#define SQ     2048
#define NB     4
#define NH     16
#define HD     128
#define HID    2048
#define QKVN   6144
#define TOKENS (SQ*NB)
#define MASKF  (-10000.0f)
#define QSCALE 0.08838834764831845f   // 1/sqrt(128)

// ---------------------------------------------------------------------------
// WMMA wrapper: D = A(16x32 bf16) * B(32x16 bf16) + C(16x16 f32)
// ---------------------------------------------------------------------------
__device__ __forceinline__ v8f wmma_bf16(v16bf a, v16bf b, v8f c) {
  return __builtin_amdgcn_wmma_f32_16x16x32_bf16(false, a, false, b,
                                                 (short)0, c, false, false);
}

// A-fragment (16x32, MxK) from row-major [16][ld] tile in LDS.
// ISA layout: lane&15 = M row; lanes 0-15 hold K {0..7,16..23}, lanes 16-31
// hold K {8..15,24..31}  -> two 16B ds loads per lane.
__device__ __forceinline__ v16bf lds_frag_a(const __bf16* tile, int ld) {
  const int lane = threadIdx.x & 31;
  const __bf16* p = tile + (lane & 15) * ld + ((lane >> 4) << 3);
  union { v16bf v; uint4 q[2]; } u;
  u.q[0] = *(const uint4*)(p);
  u.q[1] = *(const uint4*)(p + 16);
  return u.v;
}

// B-fragment (32x16, KxN) from row-major [N][ld] (i.e. NxK) tile in LDS.
// ISA layout: lane&15 = N column; lanes 0-15 hold K {0..7,16..23}, lanes 16-31
// hold K {8..15,24..31} -> two 16B ds loads per lane.
__device__ __forceinline__ v16bf lds_frag_b(const __bf16* tile, int ld) {
  const int lane = threadIdx.x & 31;
  const __bf16* p = tile + (lane & 15) * ld + ((lane >> 4) << 3);
  union { v16bf v; uint4 q[2]; } u;
  u.q[0] = *(const uint4*)(p);
  u.q[1] = *(const uint4*)(p + 16);
  return u.v;
}

// ---------------------------------------------------------------------------
// CDNA5 async global->LDS copy (ASYNCcnt-tracked, bypasses VGPRs).
// Generic->LDS offset: low 32 bits of the generic LDS pointer.
// ---------------------------------------------------------------------------
__device__ __forceinline__ void async_copy_b128(const void* gbase, unsigned voff,
                                                void* lds_ptr) {
  unsigned loff = (unsigned)(size_t)lds_ptr;
  unsigned long long base = (unsigned long long)(size_t)gbase;
  asm volatile("global_load_async_to_lds_b128 %0, %1, %2 offset:0"
               :: "v"(loff), "v"(voff), "s"(base) : "memory");
}
__device__ __forceinline__ void wait_async0() {
  asm volatile("s_wait_asynccnt 0" ::: "memory");
}

// ---------------------------------------------------------------------------
// fp32 -> bf16 elementwise convert
// ---------------------------------------------------------------------------
__global__ __launch_bounds__(256) void cvt_bf16_kernel(
    const float* __restrict__ src, __bf16* __restrict__ dst, int n) {
  int i = blockIdx.x * 256 + threadIdx.x;
  if (i < n) dst[i] = (__bf16)src[i];
}

// ---------------------------------------------------------------------------
// Tiled bf16 WMMA GEMM:  C[M,N] = A[M,K] * B[N,K]^T  (+ bias)
//   Block: 256 threads (8 waves), 128x128 output tile, BK=64.
//   Wave grid 2x4 -> each wave owns 64x32 = 4x2 WMMA tiles, 16 WMMA/iter.
//   Tiles staged with global_load_async_to_lds_b128.
// MODE 0: QKV epilogue -> scatter bf16 Q (scaled), K, V as [b][head][s][d]
// MODE 1: plain epilogue -> fp32 C += bias
// ---------------------------------------------------------------------------
template<int MODE>
__global__ __launch_bounds__(256) void gemm_bf16_kernel(
    const __bf16* __restrict__ A, const __bf16* __restrict__ B,
    const float* __restrict__ bias, float* __restrict__ Cout,
    __bf16* __restrict__ Qb, __bf16* __restrict__ Kb, __bf16* __restrict__ Vb,
    int M, int N, int Kdim) {
  __shared__ __bf16 As[128 * 72];      // 128 rows x 64 K (+8 pad)
  __shared__ __bf16 Bs[128 * 72];

  const int tid  = threadIdx.x;
  const int lane = tid & 31;
  const int wave = tid >> 5;
  const int wm   = wave >> 2;          // 0..1  (64 rows each)
  const int wn   = wave & 3;           // 0..3  (32 cols each)
  const int bm   = blockIdx.y * 128;
  const int bn   = blockIdx.x * 128;

  v8f acc[4][2];
#pragma unroll
  for (int i = 0; i < 4; ++i)
#pragma unroll
    for (int j = 0; j < 2; ++j) acc[i][j] = (v8f)0.0f;

  for (int k0 = 0; k0 < Kdim; k0 += 64) {
    __syncthreads();                   // prior iteration's reads done
#pragma unroll
    for (int j = 0; j < 4; ++j) {      // 1024 16B chunks per tile, 4/thread
      int c   = tid * 4 + j;
      int r   = c >> 3;
      int col = (c & 7) * 8;
      async_copy_b128(A, (unsigned)(((bm + r) * Kdim + k0 + col) * 2),
                      &As[r * 72 + col]);
      async_copy_b128(B, (unsigned)(((bn + r) * Kdim + k0 + col) * 2),
                      &Bs[r * 72 + col]);
    }
    wait_async0();                     // this wave's copies landed in LDS
    __syncthreads();                   // everyone's copies visible

#pragma unroll
    for (int kc = 0; kc < 2; ++kc) {
      v16bf af[4], bf[2];
#pragma unroll
      for (int ms = 0; ms < 4; ++ms)
        af[ms] = lds_frag_a(&As[(wm * 64 + ms * 16) * 72 + kc * 32], 72);
#pragma unroll
      for (int ns = 0; ns < 2; ++ns)
        bf[ns] = lds_frag_b(&Bs[(wn * 32 + ns * 16) * 72 + kc * 32], 72);
#pragma unroll
      for (int ms = 0; ms < 4; ++ms)
#pragma unroll
        for (int ns = 0; ns < 2; ++ns)
          acc[ms][ns] = wmma_bf16(af[ms], bf[ns], acc[ms][ns]);
    }
  }

  // Epilogue.  C layout: lane&15 = N col, (lane>>4)*8 + vgpr = M row.
  const int n0 = lane & 15;
  const int g  = lane >> 4;
#pragma unroll
  for (int ms = 0; ms < 4; ++ms)
#pragma unroll
    for (int ns = 0; ns < 2; ++ns)
#pragma unroll
      for (int i = 0; i < 8; ++i) {
        int gm = bm + wm * 64 + ms * 16 + g * 8 + i;
        int gn = bn + wn * 32 + ns * 16 + n0;
        float v = acc[ms][ns][i] + bias[gn];
        if (MODE == 1) {
          Cout[(size_t)gm * N + gn] = v;
        } else {
          int s    = gm >> 2;          // token = s*4 + bi
          int bi   = gm & 3;
          int head = gn / 384;
          int r    = gn - head * 384;
          size_t o = (((size_t)bi * NH + head) * SQ + s) * HD;
          if (r < 128)       Qb[o + r]       = (__bf16)(v * QSCALE);
          else if (r < 256)  Kb[o + r - 128] = (__bf16)v;
          else               Vb[o + r - 256] = (__bf16)v;
        }
      }
}

// ---------------------------------------------------------------------------
// Flash attention (causal).  grid = (b*np = 64, sq/64 = 32), 128 threads.
// Each wave owns 16 query rows; block streams 64-key K/V tiles via LDS.
// K tile staged with async global->LDS; V tile transposed through VGPRs.
// Q is pre-scaled by 1/sqrt(hn).  Only the diagonal key tile is masked;
// tiles beyond the diagonal are skipped entirely.
// ---------------------------------------------------------------------------
__global__ __launch_bounds__(128) void flash_attn_kernel(
    const __bf16* __restrict__ Qb, const __bf16* __restrict__ Kb,
    const __bf16* __restrict__ Vb, __bf16* __restrict__ ctx) {
  __shared__ __bf16 Ks[64 * 136];      // [key][d]   (NxK for QK^T B-operand)
  __shared__ __bf16 Vt[128 * 72];      // [d][key]   (NxK for PV  B-operand)
  __shared__ __bf16 Ps[4][16 * 72];    // per-wave P tile [m][key]

  const int hb   = blockIdx.x;         // bi*16 + head
  const int qt   = blockIdx.y;         // 64-row query tile
  const int tid  = threadIdx.x;
  const int lane = tid & 31;
  const int wave = tid >> 5;
  const int n0   = lane & 15;
  const int g    = lane >> 4;

  const __bf16* Qh = Qb + (size_t)hb * SQ * HD;
  const __bf16* Kh = Kb + (size_t)hb * SQ * HD;
  const __bf16* Vh = Vb + (size_t)hb * SQ * HD;

  // Q fragments for this wave's 16 rows: 4 K-chunks of 32, held in regs.
  v16bf qf[4];
  {
    const __bf16* qp = Qh + (size_t)(qt * 64 + wave * 16 + n0) * HD;
#pragma unroll
    for (int kc = 0; kc < 4; ++kc) {
      union { v16bf v; uint4 q[2]; } u;
      u.q[0] = *(const uint4*)(qp + kc * 32 + g * 8);
      u.q[1] = *(const uint4*)(qp + kc * 32 + g * 8 + 16);
      qf[kc] = u.v;
    }
  }

  v8f O[8];
  float mrow[8], lrow[8];
#pragma unroll
  for (int i = 0; i < 8; ++i) { O[i] = (v8f)0.0f; mrow[i] = -3.0e38f; lrow[i] = 0.0f; }

  for (int kt = 0; kt <= qt; ++kt) {
    __syncthreads();                   // prior iteration's LDS reads done
    // Stage K tile [64][128] row-major via async copy.
#pragma unroll
    for (int c = tid; c < 1024; c += 128) {
      int r = c >> 4, col = (c & 15) * 8;
      async_copy_b128(Kh, (unsigned)(((kt * 64 + r) * HD + col) * 2),
                      &Ks[r * 136 + col]);
    }
    // Stage V tile transposed -> Vt[d][key] (element transpose via VGPRs).
#pragma unroll
    for (int c = tid; c < 1024; c += 128) {
      int r = c >> 4, col = (c & 15) * 8;
      union { uint4 q; __bf16 e[8]; } u;
      u.q = *(const uint4*)&Vh[((size_t)kt * 64 + r) * HD + col];
#pragma unroll
      for (int j = 0; j < 8; ++j) Vt[(col + j) * 72 + r] = u.e[j];
    }
    wait_async0();
    __syncthreads();

    // S = Q * K^T : 16x64 per wave (4 N-subtiles x 4 K-chunks).
    v8f S[4];
#pragma unroll
    for (int ns = 0; ns < 4; ++ns) {
      v8f s = (v8f)0.0f;
#pragma unroll
      for (int kc = 0; kc < 4; ++kc)
        s = wmma_bf16(qf[kc], lds_frag_b(&Ks[(ns * 16) * 136 + kc * 32], 136), s);
      S[ns] = s;
    }

    // Causal mask, diagonal tile only (qt*64 cancels on both sides).
    if (kt == qt) {
#pragma unroll
      for (int ns = 0; ns < 4; ++ns)
#pragma unroll
        for (int i = 0; i < 8; ++i) {
          int qs = wave * 16 + g * 8 + i;
          int ks = ns * 16 + n0;
          if (ks > qs) S[ns][i] = MASKF;
        }
    }

    // Online softmax.  Row m = g*8+i lives in one 16-lane half -> butterfly
    // over xor masks 1,2,4,8 stays inside the half (wave32).
    float tmax[8];
#pragma unroll
    for (int i = 0; i < 8; ++i)
      tmax[i] = fmaxf(fmaxf(S[0][i], S[1][i]), fmaxf(S[2][i], S[3][i]));
#pragma unroll
    for (int off = 1; off < 16; off <<= 1)
#pragma unroll
      for (int i = 0; i < 8; ++i)
        tmax[i] = fmaxf(tmax[i], __shfl_xor(tmax[i], off, 32));

    float alpha[8];
#pragma unroll
    for (int i = 0; i < 8; ++i) {
      float mnew = fmaxf(mrow[i], tmax[i]);
      alpha[i] = __expf(mrow[i] - mnew);
      mrow[i] = mnew;
    }
#pragma unroll
    for (int ns = 0; ns < 4; ++ns)
#pragma unroll
      for (int i = 0; i < 8; ++i)
        S[ns][i] = __expf(S[ns][i] - mrow[i]);

    float rs[8];
#pragma unroll
    for (int i = 0; i < 8; ++i)
      rs[i] = (S[0][i] + S[1][i]) + (S[2][i] + S[3][i]);
#pragma unroll
    for (int off = 1; off < 16; off <<= 1)
#pragma unroll
      for (int i = 0; i < 8; ++i)
        rs[i] += __shfl_xor(rs[i], off, 32);
#pragma unroll
    for (int i = 0; i < 8; ++i) lrow[i] = lrow[i] * alpha[i] + rs[i];
#pragma unroll
    for (int d8 = 0; d8 < 8; ++d8)
#pragma unroll
      for (int i = 0; i < 8; ++i) O[d8][i] *= alpha[i];

    // Re-layout P (C-layout -> A-layout) through per-wave LDS scratch.
    __bf16* Pw = &Ps[wave][0];
#pragma unroll
    for (int ns = 0; ns < 4; ++ns)
#pragma unroll
      for (int i = 0; i < 8; ++i)
        Pw[(g * 8 + i) * 72 + ns * 16 + n0] = (__bf16)S[ns][i];
    // Same-wave LDS ops are in-order (DScnt): store->load safe.

    v16bf pa0 = lds_frag_a(Pw, 72);        // keys  0..31
    v16bf pa1 = lds_frag_a(Pw + 32, 72);   // keys 32..63
#pragma unroll
    for (int d8 = 0; d8 < 8; ++d8) {
      v8f o = O[d8];
      o = wmma_bf16(pa0, lds_frag_b(&Vt[(d8 * 16) * 72],      72), o);
      o = wmma_bf16(pa1, lds_frag_b(&Vt[(d8 * 16) * 72 + 32], 72), o);
      O[d8] = o;
    }
  }

  // Normalize and scatter ctx as bf16 [token][head*128+d].
  const int head = hb & 15, bi = hb >> 4;
  float rl[8];
#pragma unroll
  for (int i = 0; i < 8; ++i) rl[i] = 1.0f / lrow[i];
#pragma unroll
  for (int d8 = 0; d8 < 8; ++d8)
#pragma unroll
    for (int i = 0; i < 8; ++i) {
      int s = qt * 64 + wave * 16 + g * 8 + i;
      int d = d8 * 16 + n0;
      ctx[((size_t)s * 4 + bi) * HID + head * 128 + d] = (__bf16)(O[d8][i] * rl[i]);
    }
}

// ---------------------------------------------------------------------------
// Host-side orchestration
// ---------------------------------------------------------------------------
extern "C" void kernel_launch(void* const* d_in, const int* in_sizes, int n_in,
                              void* d_out, int out_size, void* d_ws, size_t ws_size,
                              hipStream_t stream) {
  const float* hs      = (const float*)d_in[0];
  // d_in[1]: attention_mask (causal) — applied analytically in flash kernel.
  const float* w_qkv   = (const float*)d_in[2];
  const float* b_qkv   = (const float*)d_in[3];
  const float* w_dense = (const float*)d_in[4];
  const float* b_dense = (const float*)d_in[5];
  float* out = (float*)d_out;

  char* p = (char*)d_ws;
  __bf16* hsb   = (__bf16*)p; p += (size_t)TOKENS * HID * 2;   // also reused as ctx
  __bf16* wqkvb = (__bf16*)p; p += (size_t)QKVN * HID * 2;
  __bf16* wdb   = (__bf16*)p; p += (size_t)HID * HID * 2;
  __bf16* Qb    = (__bf16*)p; p += (size_t)NB * NH * SQ * HD * 2;
  __bf16* Kbuf  = (__bf16*)p; p += (size_t)NB * NH * SQ * HD * 2;
  __bf16* Vbuf  = (__bf16*)p; p += (size_t)NB * NH * SQ * HD * 2;
  __bf16* ctxb  = hsb;  // hidden bf16 dead after QKV GEMM -> alias for ctx

  // 1) fp32 -> bf16 converts
  {
    int n = TOKENS * HID;
    cvt_bf16_kernel<<<(n + 255) / 256, 256, 0, stream>>>(hs, hsb, n);
    n = QKVN * HID;
    cvt_bf16_kernel<<<(n + 255) / 256, 256, 0, stream>>>(w_qkv, wqkvb, n);
    n = HID * HID;
    cvt_bf16_kernel<<<(n + 255) / 256, 256, 0, stream>>>(w_dense, wdb, n);
  }

  // 2) QKV projection + head scatter (Q pre-scaled by 1/sqrt(hn))
  gemm_bf16_kernel<0><<<dim3(QKVN / 128, TOKENS / 128), 256, 0, stream>>>(
      hsb, wqkvb, b_qkv, nullptr, Qb, Kbuf, Vbuf, TOKENS, QKVN, HID);

  // 3) causal flash attention -> ctx bf16
  flash_attn_kernel<<<dim3(NB * NH, SQ / 64), 128, 0, stream>>>(Qb, Kbuf, Vbuf, ctxb);

  // 4) output projection + bias -> fp32 out
  gemm_bf16_kernel<1><<<dim3(HID / 128, TOKENS / 128), 256, 0, stream>>>(
      ctxb, wdb, b_dense, out, nullptr, nullptr, nullptr, TOKENS, HID, HID);
}
